// LinearAttentionBlock_52063593562953
// MI455X (gfx1250) — matrix operand order repaired and checked by reading, not compiled, not run on hardware
//
#include <hip/hip_runtime.h>
#include <hip/hip_bf16.h>

// LinearAttentionBlock for MI455X (gfx1250): bf16 WMMA GEMM pipeline with
// TDM (tensor_load_to_lds) double-buffered A-tile staging.
// B=4, S=4096, D=1024, H=16, DH=64, FF=4096.

#define B_   4
#define S_   4096
#define D_   1024
#define H_   16
#define DH_  64
#define FF_  4096
#define NTOK (B_ * S_)

typedef __attribute__((ext_vector_type(16))) __bf16 v16bf;
typedef __attribute__((ext_vector_type(8)))  float  v8f;
typedef __attribute__((ext_vector_type(4)))  unsigned int u32x4;
typedef __attribute__((ext_vector_type(8)))  int          i32x8;

union FragB {
    v16bf v;
    uint4 q[2];
};

__device__ __forceinline__ unsigned short f2bf(float f) {
    unsigned u = __builtin_bit_cast(unsigned, f);
    u += 0x7FFFu + ((u >> 16) & 1u);          // round-to-nearest-even
    return (unsigned short)(u >> 16);
}
__device__ __forceinline__ float bf2f(unsigned short h) {
    return __builtin_bit_cast(float, (unsigned)h << 16);
}

// Issue a TDM load of a [128 rows x 32 cols] bf16 tile from a row-major
// [M x K] tensor (element strides) into LDS at byte offset ldsOff.
// D# layout per CDNA5 ISA (2-group, 2-D form).
__device__ __forceinline__ void tdm_load_tileA(const unsigned short* gtile,
                                               unsigned ldsOff, int K, int M) {
    unsigned long long ga = (unsigned long long)(uintptr_t)gtile;
    u32x4 g0;
    g0[0] = 1u;                                   // count=1, user mode
    g0[1] = ldsOff;                               // lds_addr (bytes)
    g0[2] = (unsigned)(ga & 0xFFFFFFFFu);         // global_addr[31:0]
    g0[3] = (unsigned)((ga >> 32) & 0x1FFFFFFu)   // global_addr[56:32]
            | 0x80000000u;                        // type=2 ("image")
    i32x8 g1;
    g1[0] = 0x00010000;                           // data_size=1 (2 bytes)
    g1[1] = (int)(((unsigned)K & 0xFFFFu) << 16);                 // tensor_dim0 lo
    g1[2] = (int)(((unsigned)K >> 16) | (((unsigned)M & 0xFFFFu) << 16));
    g1[3] = (int)(((unsigned)M >> 16) | (32u << 16));             // tile_dim0=32
    g1[4] = 128;                                  // tile_dim1=128, tile_dim2=0
    g1[5] = K;                                    // tensor_dim0_stride lo32
    g1[6] = 0;                                    // stride hi, dim1_stride lo
    g1[7] = 0;
    asm volatile("tensor_load_to_lds %0, %1" ::"s"(g0), "s"(g1) : "memory");
}

// ---------------------------------------------------------------- f32 -> bf16
__global__ __launch_bounds__(256) void cvt_f32_bf16(const float* __restrict__ in,
                                                    unsigned short* __restrict__ out,
                                                    int n) {
    int i = blockIdx.x * 256 + threadIdx.x;
    if (i < n) out[i] = f2bf(in[i]);
}

// ---------------------------------------------------------------- generic GEMM
// C[M,N] = A[M,K](bf16) * B[K,N](bf16) + bias, fused epilogue by MODE.
// Block tile 128x64, 8 waves, each wave: 32x32 (2x2 WMMA 16x16x32 bf16 tiles).
// A tiles: TDM double-buffered into LDS; B tiles: pipelined load + transpose.
enum { M_ELU = 0, M_BF16 = 1, M_RESID = 2, M_GELU = 3, M_ADDF = 4 };

template <int MODE>
__global__ __launch_bounds__(256) void gemm_wmma_bf16(
    const unsigned short* __restrict__ A,    // [M,K]
    const unsigned short* __restrict__ Bw,   // [K,N]
    const float* __restrict__ bias,          // [N]
    const float* __restrict__ resid,         // [M,N] (M_RESID)
    float* __restrict__ outF,                // [M,N] (M_RESID / M_ADDF)
    unsigned short* __restrict__ outB,       // [M,N] (bf16 modes)
    int M, int N, int K) {
    __shared__ unsigned short ldsA[2][128 * 32];  // row-major A tiles (TDM dest)
    __shared__ unsigned short ldsB[2][64 * 32];   // transposed B tiles [n][k]

    const int tid  = threadIdx.x;
    const int wave = tid >> 5;
    const int lane = tid & 31;
    const int half = lane >> 4;
    const int lr   = lane & 15;
    const int wm   = wave & 3;   // 32-row band inside block tile
    const int wn   = wave >> 2;  // 32-col band
    const int m0   = blockIdx.y * 128;
    const int n0   = blockIdx.x * 64;

    const unsigned ldsOffA[2] = {(unsigned)(uintptr_t)&ldsA[0][0],
                                 (unsigned)(uintptr_t)&ldsA[1][0]};

    v8f acc[2][2];
    acc[0][0] = {}; acc[0][1] = {}; acc[1][0] = {}; acc[1][1] = {};

    const int brow = tid >> 3, bcg = (tid & 7) * 8;  // 32 k-rows x 8 col-chunks

    // ---- prologue: chunk 0 into buffer 0
    if (wave == 0) tdm_load_tileA(A + (size_t)m0 * K, ldsOffA[0], K, M);
    {
        uint4 bv = *(const uint4*)(Bw + (size_t)brow * N + n0 + bcg);
        const unsigned short* bs = (const unsigned short*)&bv;
#pragma unroll
        for (int j = 0; j < 8; ++j) ldsB[0][(bcg + j) * 32 + brow] = bs[j];
    }
    if (wave == 0) __builtin_amdgcn_s_wait_tensorcnt(0);
    __syncthreads();

    for (int kc = 0; kc < K; kc += 32) {
        const int buf = (kc >> 5) & 1;
        const bool more = (kc + 32) < K;
        uint4 bv;
        if (more) {
            // kick next A tile via TDM into the idle buffer; pull next B chunk
            if (wave == 0)
                tdm_load_tileA(A + (size_t)m0 * K + kc + 32, ldsOffA[1 - buf], K, M);
            bv = *(const uint4*)(Bw + (size_t)(kc + 32 + brow) * N + n0 + bcg);
        }

        FragB af[2], bfr[2];
#pragma unroll
        for (int mt = 0; mt < 2; ++mt) {
            int m = wm * 32 + mt * 16 + lr;
            af[mt].q[0] = *(const uint4*)(&ldsA[buf][m * 32 + half * 16]);
            af[mt].q[1] = *(const uint4*)(&ldsA[buf][m * 32 + half * 16 + 8]);
        }
#pragma unroll
        for (int nt = 0; nt < 2; ++nt) {
            int n = wn * 32 + nt * 16 + lr;
            bfr[nt].q[0] = *(const uint4*)(&ldsB[buf][n * 32 + half * 16]);
            bfr[nt].q[1] = *(const uint4*)(&ldsB[buf][n * 32 + half * 16 + 8]);
        }
#pragma unroll
        for (int mt = 0; mt < 2; ++mt)
#pragma unroll
            for (int nt = 0; nt < 2; ++nt)
                acc[mt][nt] = __builtin_amdgcn_wmma_f32_16x16x32_bf16(
                    false, af[mt].v, false, bfr[nt].v, (short)0, acc[mt][nt],
                    false, false);

        if (more) {
            // scatter next B chunk transposed into the idle buffer
            const unsigned short* bs = (const unsigned short*)&bv;
#pragma unroll
            for (int j = 0; j < 8; ++j) ldsB[1 - buf][(bcg + j) * 32 + brow] = bs[j];
            if (wave == 0) __builtin_amdgcn_s_wait_tensorcnt(0);
        }
        __syncthreads();
    }

#pragma unroll
    for (int mt = 0; mt < 2; ++mt)
#pragma unroll
        for (int nt = 0; nt < 2; ++nt)
#pragma unroll
            for (int r = 0; r < 8; ++r) {
                int row = m0 + wm * 32 + mt * 16 + half * 8 + r;
                int col = n0 + wn * 32 + nt * 16 + lr;
                float v = acc[mt][nt][r] + bias[col];
                size_t idx = (size_t)row * N + col;
                if (MODE == M_ELU) {               // phi(x)=elu(x)+1
                    v = v > 0.f ? v + 1.f : __expf(v);
                    outB[idx] = f2bf(v);
                } else if (MODE == M_BF16) {
                    outB[idx] = f2bf(v);
                } else if (MODE == M_RESID) {
                    outF[idx] = v + resid[idx];
                } else if (MODE == M_GELU) {       // exact erf gelu
                    v = 0.5f * v * (1.f + erff(v * 0.70710678118f));
                    outB[idx] = f2bf(v);
                } else {                           // M_ADDF
                    outF[idx] = v;
                }
            }
}

// ---------------------------------------------------------------- KV = K^T V
// Per (b,h): [DH=64, S=4096] x [S, DH=64]. Output stored transposed bf16:
// kvT[bh][m][d] so the o-GEMM can read B fragments contiguously.
__global__ __launch_bounds__(256) void kv_wmma(const unsigned short* __restrict__ Kb,
                                               const unsigned short* __restrict__ Vb,
                                               unsigned short* __restrict__ kvT) {
    __shared__ unsigned short ldsK[64 * 32];  // [d][s]
    __shared__ unsigned short ldsV[64 * 32];  // [m][s]
    const int bh = blockIdx.x, b = bh >> 4, h = bh & 15;
    const int tid = threadIdx.x, wave = tid >> 5, lane = tid & 31;
    const int half = lane >> 4, lr = lane & 15;
    const int wm = wave & 3;    // d tile (4 x 16)
    const int wn = wave >> 2;   // m tile pair (2 x 2x16)
    v8f acc[2]; acc[0] = {}; acc[1] = {};
    const int srow = tid >> 3, dg = (tid & 7) * 8;
    const size_t base = (size_t)b * S_ * (H_ * DH_) + h * DH_;

    for (int s0 = 0; s0 < S_; s0 += 32) {
        uint4 kq = *(const uint4*)(Kb + base + (size_t)(s0 + srow) * (H_ * DH_) + dg);
        uint4 vq = *(const uint4*)(Vb + base + (size_t)(s0 + srow) * (H_ * DH_) + dg);
        const unsigned short* kp = (const unsigned short*)&kq;
        const unsigned short* vp = (const unsigned short*)&vq;
#pragma unroll
        for (int j = 0; j < 8; ++j) {
            ldsK[(dg + j) * 32 + srow] = kp[j];
            ldsV[(dg + j) * 32 + srow] = vp[j];
        }
        __syncthreads();

        FragB af, b0, b1;
        int d = wm * 16 + lr;
        af.q[0] = *(const uint4*)(&ldsK[d * 32 + half * 16]);
        af.q[1] = *(const uint4*)(&ldsK[d * 32 + half * 16 + 8]);
        int n0c = (wn * 2) * 16 + lr;
        b0.q[0] = *(const uint4*)(&ldsV[n0c * 32 + half * 16]);
        b0.q[1] = *(const uint4*)(&ldsV[n0c * 32 + half * 16 + 8]);
        int n1c = (wn * 2 + 1) * 16 + lr;
        b1.q[0] = *(const uint4*)(&ldsV[n1c * 32 + half * 16]);
        b1.q[1] = *(const uint4*)(&ldsV[n1c * 32 + half * 16 + 8]);
        acc[0] = __builtin_amdgcn_wmma_f32_16x16x32_bf16(false, af.v, false, b0.v,
                                                         (short)0, acc[0], false, false);
        acc[1] = __builtin_amdgcn_wmma_f32_16x16x32_bf16(false, af.v, false, b1.v,
                                                         (short)0, acc[1], false, false);
        __syncthreads();
    }
#pragma unroll
    for (int nt = 0; nt < 2; ++nt)
#pragma unroll
        for (int r = 0; r < 8; ++r) {
            int d = wm * 16 + half * 8 + r;
            int m = (wn * 2 + nt) * 16 + lr;
            kvT[(size_t)bh * 4096 + m * 64 + d] = f2bf(acc[nt][r]);
        }
}

// ---------------------------------------------------------------- Ksum, Z
__global__ __launch_bounds__(64) void ksum_kernel(const unsigned short* __restrict__ Kb,
                                                  float* __restrict__ Ksum) {
    int bh = blockIdx.x, b = bh >> 4, h = bh & 15, d = threadIdx.x;
    const unsigned short* p = Kb + (size_t)b * S_ * (H_ * DH_) + h * DH_ + d;
    float s = 0.f;
    for (int i = 0; i < S_; ++i) s += bf2f(p[(size_t)i * (H_ * DH_)]);
    Ksum[bh * 64 + d] = s;
}

__global__ __launch_bounds__(256) void z_kernel(const unsigned short* __restrict__ Qb,
                                                const float* __restrict__ Ksum,
                                                float* __restrict__ Z) {
    int gid = blockIdx.x * 256 + threadIdx.x;  // (b*S+s)*H + h
    int h = gid & 15, bs = gid >> 4, b = bs >> 12;
    const unsigned short* q = Qb + (size_t)bs * (H_ * DH_) + h * DH_;
    const float* ks = Ksum + (b * 16 + h) * 64;
    float dot = 0.f;
    for (int d = 0; d < 64; ++d) dot += bf2f(q[d]) * ks[d];
    Z[gid] = 1.f / (dot + 1e-6f);
}

// ---------------------------------------------------------------- o = Z*(Q@KV)
// Per (b,h): [S,64] x [64,64]; A and B fragments loaded straight from global.
__global__ __launch_bounds__(256) void o_wmma(const unsigned short* __restrict__ Qb,
                                              const unsigned short* __restrict__ kvT,
                                              const float* __restrict__ Z,
                                              unsigned short* __restrict__ oB) {
    const int bh = blockIdx.y, b = bh >> 4, h = bh & 15;
    const int s0 = blockIdx.x * 128;
    const int tid = threadIdx.x, wave = tid >> 5, lane = tid & 31;
    const int half = lane >> 4, lr = lane & 15;
    const int wm = wave & 3, wn = wave >> 2;
    v8f acc[2][2];
    acc[0][0] = {}; acc[0][1] = {}; acc[1][0] = {}; acc[1][1] = {};
    const unsigned short* qbase = Qb + (size_t)b * S_ * (H_ * DH_) + h * DH_;
    const unsigned short* kvb   = kvT + (size_t)bh * 4096;

#pragma unroll
    for (int kc = 0; kc < 64; kc += 32) {
        FragB af[2], bfr[2];
#pragma unroll
        for (int mt = 0; mt < 2; ++mt) {
            int srow = s0 + wm * 32 + mt * 16 + lr;
            const unsigned short* p = qbase + (size_t)srow * (H_ * DH_) + kc;
            af[mt].q[0] = *(const uint4*)(p + half * 16);
            af[mt].q[1] = *(const uint4*)(p + half * 16 + 8);
        }
#pragma unroll
        for (int nt = 0; nt < 2; ++nt) {
            int mcol = wn * 32 + nt * 16 + lr;
            const unsigned short* p = kvb + mcol * 64 + kc;
            bfr[nt].q[0] = *(const uint4*)(p + half * 16);
            bfr[nt].q[1] = *(const uint4*)(p + half * 16 + 8);
        }
#pragma unroll
        for (int mt = 0; mt < 2; ++mt)
#pragma unroll
            for (int nt = 0; nt < 2; ++nt)
                acc[mt][nt] = __builtin_amdgcn_wmma_f32_16x16x32_bf16(
                    false, af[mt].v, false, bfr[nt].v, (short)0, acc[mt][nt],
                    false, false);
    }
#pragma unroll
    for (int mt = 0; mt < 2; ++mt)
#pragma unroll
        for (int nt = 0; nt < 2; ++nt)
#pragma unroll
            for (int r = 0; r < 8; ++r) {
                int srow = s0 + wm * 32 + mt * 16 + half * 8 + r;
                int mcol = wn * 32 + nt * 16 + lr;
                float z = Z[((size_t)b * S_ + srow) * H_ + h];
                oB[((size_t)b * S_ + srow) * (H_ * DH_) + h * DH_ + mcol] =
                    f2bf(acc[mt][nt][r] * z);
            }
}

// ---------------------------------------------------------------- LayerNorm
// FUSE=0: x=LN(X), write f32 + bf16.  FUSE=1: out=LN(X+Y), write f32 only.
template <int FUSE>
__global__ __launch_bounds__(256) void ln_kernel(const float* __restrict__ X,
                                                 const float* __restrict__ Y,
                                                 const float* __restrict__ g,
                                                 const float* __restrict__ bta,
                                                 float* __restrict__ outF,
                                                 unsigned short* __restrict__ outB) {
    __shared__ float r1[256], r2[256];
    const int row = blockIdx.x, tid = threadIdx.x;
    const size_t base = (size_t)row * D_;
    float vals[4], s1 = 0.f, s2 = 0.f;
#pragma unroll
    for (int j = 0; j < 4; ++j) {
        int i = tid + j * 256;
        float v = X[base + i];
        if (FUSE) v += Y[base + i];
        vals[j] = v; s1 += v; s2 += v * v;
    }
    r1[tid] = s1; r2[tid] = s2; __syncthreads();
    for (int st = 128; st > 0; st >>= 1) {
        if (tid < st) { r1[tid] += r1[tid + st]; r2[tid] += r2[tid + st]; }
        __syncthreads();
    }
    float mean = r1[0] * (1.f / D_);
    float var  = r2[0] * (1.f / D_) - mean * mean;
    float rstd = rsqrtf(var + 1e-5f);
#pragma unroll
    for (int j = 0; j < 4; ++j) {
        int i = tid + j * 256;
        float o = (vals[j] - mean) * rstd * g[i] + bta[i];
        outF[base + i] = o;
        if (!FUSE) outB[base + i] = f2bf(o);
    }
}

// ---------------------------------------------------------------- launcher
extern "C" void kernel_launch(void* const* d_in, const int* in_sizes, int n_in,
                              void* d_out, int out_size, void* d_ws, size_t ws_size,
                              hipStream_t stream) {
    (void)in_sizes; (void)n_in; (void)out_size; (void)ws_size;
    const float* src = (const float*)d_in[0];
    const float* Wq  = (const float*)d_in[1];
    const float* bq  = (const float*)d_in[2];
    const float* Wk  = (const float*)d_in[3];
    const float* bk  = (const float*)d_in[4];
    const float* Wv  = (const float*)d_in[5];
    const float* bv  = (const float*)d_in[6];
    const float* Wo  = (const float*)d_in[7];
    const float* bo  = (const float*)d_in[8];
    const float* W1  = (const float*)d_in[9];
    const float* b1  = (const float*)d_in[10];
    const float* W2  = (const float*)d_in[11];
    const float* b2  = (const float*)d_in[12];
    const float* g1  = (const float*)d_in[13];
    const float* be1 = (const float*)d_in[14];
    const float* g2  = (const float*)d_in[15];
    const float* be2 = (const float*)d_in[16];
    float* out = (float*)d_out;

    char* w = (char*)d_ws; size_t off = 0;
    auto carve = [&](size_t bytes) -> void* {
        void* p = w + off;
        off += (bytes + 255) & ~(size_t)255;
        return p;
    };
    unsigned short* srcB = (unsigned short*)carve((size_t)NTOK * D_ * 2);   // later x_bf16
    unsigned short* WqB  = (unsigned short*)carve((size_t)D_ * 1024 * 2);
    unsigned short* WkB  = (unsigned short*)carve((size_t)D_ * 1024 * 2);
    unsigned short* WvB  = (unsigned short*)carve((size_t)D_ * 1024 * 2);
    unsigned short* WoB  = (unsigned short*)carve((size_t)D_ * 1024 * 2);
    unsigned short* W1B  = (unsigned short*)carve((size_t)D_ * FF_ * 2);
    unsigned short* W2B  = (unsigned short*)carve((size_t)FF_ * D_ * 2);
    unsigned short* Qb   = (unsigned short*)carve((size_t)NTOK * 1024 * 2);
    unsigned short* Kb   = (unsigned short*)carve((size_t)NTOK * 1024 * 2); // later o
    unsigned short* Vb   = (unsigned short*)carve((size_t)NTOK * 1024 * 2);
    unsigned short* kvT  = (unsigned short*)carve((size_t)64 * 4096 * 2);
    float* Ksum = (float*)carve((size_t)64 * 64 * 4);
    float* Zbuf = (float*)carve((size_t)NTOK * H_ * 4);
    float* xF   = (float*)carve((size_t)NTOK * D_ * 4);
    unsigned short* hB = (unsigned short*)carve((size_t)NTOK * FF_ * 2);

    // 1) precision conversion (src + weights)
    int nSrc = NTOK * D_;
    cvt_f32_bf16<<<(nSrc + 255) / 256, 256, 0, stream>>>(src, srcB, nSrc);
    cvt_f32_bf16<<<(D_ * 1024 + 255) / 256, 256, 0, stream>>>(Wq, WqB, D_ * 1024);
    cvt_f32_bf16<<<(D_ * 1024 + 255) / 256, 256, 0, stream>>>(Wk, WkB, D_ * 1024);
    cvt_f32_bf16<<<(D_ * 1024 + 255) / 256, 256, 0, stream>>>(Wv, WvB, D_ * 1024);
    cvt_f32_bf16<<<(D_ * 1024 + 255) / 256, 256, 0, stream>>>(Wo, WoB, D_ * 1024);
    cvt_f32_bf16<<<(D_ * FF_ + 255) / 256, 256, 0, stream>>>(W1, W1B, D_ * FF_);
    cvt_f32_bf16<<<(FF_ * D_ + 255) / 256, 256, 0, stream>>>(W2, W2B, FF_ * D_);

    // 2) Q/K/V projections with fused phi / bias
    dim3 gProj(1024 / 64, NTOK / 128);
    gemm_wmma_bf16<M_ELU><<<gProj, 256, 0, stream>>>(srcB, WqB, bq, nullptr, nullptr, Qb,
                                                     NTOK, 1024, 1024);
    gemm_wmma_bf16<M_ELU><<<gProj, 256, 0, stream>>>(srcB, WkB, bk, nullptr, nullptr, Kb,
                                                     NTOK, 1024, 1024);
    gemm_wmma_bf16<M_BF16><<<gProj, 256, 0, stream>>>(srcB, WvB, bv, nullptr, nullptr, Vb,
                                                      NTOK, 1024, 1024);

    // 3) linear attention core
    kv_wmma<<<B_ * H_, 256, 0, stream>>>(Kb, Vb, kvT);
    ksum_kernel<<<B_ * H_, 64, 0, stream>>>(Kb, Ksum);
    z_kernel<<<(NTOK * H_) / 256, 256, 0, stream>>>(Qb, Ksum, Zbuf);
    o_wmma<<<dim3(S_ / 128, B_ * H_), 256, 0, stream>>>(Qb, kvT, Zbuf, Kb);  // o -> Kb

    // 4) output projection + residual, LN1 (emits x f32 in-place + x bf16 in srcB)
    gemm_wmma_bf16<M_RESID><<<gProj, 256, 0, stream>>>(Kb, WoB, bo, src, xF, nullptr,
                                                       NTOK, 1024, 1024);
    ln_kernel<0><<<NTOK, 256, 0, stream>>>(xF, nullptr, g1, be1, xF, srcB);

    // 5) FFN: gelu(x@W1+b1)@W2+b2, then LN2 of (x + y)
    dim3 gFF1(FF_ / 64, NTOK / 128);
    gemm_wmma_bf16<M_GELU><<<gFF1, 256, 0, stream>>>(srcB, W1B, b1, nullptr, nullptr, hB,
                                                     NTOK, FF_, 1024);
    gemm_wmma_bf16<M_ADDF><<<gProj, 256, 0, stream>>>(hB, W2B, b2, nullptr, out, nullptr,
                                                      NTOK, 1024, FF_);
    ln_kernel<1><<<NTOK, 256, 0, stream>>>(xF, out, g2, be2, out, nullptr);
}
